// LayerStacks_42795054137555
// MI455X (gfx1250) — matrix operand order repaired and checked by reading, not compiled, not run on hardware
//
#include <hip/hip_runtime.h>

// LayerStacks fused kernel for gfx1250 (MI455X).
// Stage-1 GEMM (16384x3072 @ 3072x144) via v_wmma_f32_16x16x32_f16,
// fused per-row MLP epilogue. x staged to LDS with NT async global->LDS copies.

#define L1DIM   3072
#define NCOL    144          // 128 stacked cols + 16 factorizer cols
#define BATCH   16384
#define RPB     64           // rows per block
#define NTHREADS 128         // 4 waves (wave32)
#define NWAVES  4
#define KC      64           // staged K chunk == 2 WMMA k-tiles
#define NSTEPS  (L1DIM / KC) // 48
#define XSTRIDE 68           // padded f32 row stride in LDS (16B aligned, banks 4m mod 64 distinct)
#define ACCSTRIDE 148        // padded f32 stride for epilogue tile
#define NTILES  9            // 144 / 16
#define CPOL_NT 1            // temporal hint: non-temporal (protect L2 residency of weights)

typedef __attribute__((ext_vector_type(16))) _Float16 v16h;
typedef __attribute__((ext_vector_type(8)))  float    v8f;
typedef __attribute__((ext_vector_type(4)))  float    v4f;
typedef __attribute__((ext_vector_type(4)))  int      v4i;

#if __has_builtin(__builtin_amdgcn_global_load_async_to_lds_b128)
#define HAVE_ASYNC 1
#endif

// Builtin expects: (v4i __device__* src, v4i __shared__* dst, imm offset, imm cpol)
typedef __attribute__((address_space(1))) v4i* gptr_t;
typedef __attribute__((address_space(3))) v4i* lptr_t;

__device__ __forceinline__ gptr_t to_gas(const void* p) {
  return (gptr_t)(unsigned long long)p;
}
__device__ __forceinline__ lptr_t to_lds(void* p) {
  // generic shared ptr: low 32 bits are the LDS offset (aperture lives in high bits)
  return (lptr_t)(unsigned long long)(unsigned int)(unsigned long long)p;
}

__device__ __forceinline__ void cp16(const float* g, float* l) {
#ifdef HAVE_ASYNC
  __builtin_amdgcn_global_load_async_to_lds_b128(to_gas(g), to_lds(l), 0, CPOL_NT);
#else
  *(v4f*)l = *(const v4f*)g;
#endif
}

template <int N>
__device__ __forceinline__ void wait_async() {
#ifdef HAVE_ASYNC
#if __has_builtin(__builtin_amdgcn_s_wait_asynccnt)
  __builtin_amdgcn_s_wait_asynccnt(N);
#else
  asm volatile("s_wait_asynccnt %0" ::"i"(N) : "memory");
#endif
#endif
}

// ---------------- pre-kernel: pack [l1_w ; l1f_w] -> f16 [144][3072] ----------------
__global__ void convert_w(const float* __restrict__ l1_w,
                          const float* __restrict__ l1f_w,
                          _Float16* __restrict__ wf16) {
  const int stride = gridDim.x * blockDim.x;
  for (int i = blockIdx.x * blockDim.x + threadIdx.x; i < NCOL * L1DIM; i += stride) {
    const int row = i / L1DIM;
    const int k   = i - row * L1DIM;
    const float v = (row < 128) ? l1_w[i] : l1f_w[(row - 128) * L1DIM + k];
    wf16[i] = (_Float16)v;
  }
}

// ---------------- fused main kernel ----------------
__global__ __launch_bounds__(NTHREADS) void
layerstacks_fused(const float* __restrict__ x,
                  const int* __restrict__ ls_indices,
                  const _Float16* __restrict__ wf16,
                  const float* __restrict__ l1_b,
                  const float* __restrict__ l1f_b,
                  const float* __restrict__ l2_w,
                  const float* __restrict__ l2_b,
                  const float* __restrict__ out_w,
                  const float* __restrict__ out_b,
                  float* __restrict__ out) {
  __shared__ __align__(16) float xbuf[2][RPB * XSTRIDE];              // 34.8 KB
  __shared__ __align__(16) float accbuf[NWAVES * 16 * ACCSTRIDE];     // 37.9 KB
  __shared__ float s_l2w[256 * 30];                                   // 30.7 KB
  __shared__ float s_l2b[256];
  __shared__ float s_l1b[128];
  __shared__ float s_l1fb[16];
  __shared__ float s_outw[256];
  __shared__ float s_outb[8];

  const int tid  = threadIdx.x;
  const int wave = tid >> 5;
  const int lane = tid & 31;
  const int m    = lane & 15;   // A: M row / B: N col within tile
  const int half = lane >> 4;
  const int rowBase = blockIdx.x * RPB;

  // Stage small stage-2/3 params into LDS (visible after first loop barrier).
  for (int i = tid; i < 256 * 30; i += NTHREADS) s_l2w[i] = l2_w[i];
  for (int i = tid; i < 256; i += NTHREADS) { s_l2b[i] = l2_b[i]; s_outw[i] = out_w[i]; }
  if (tid < 128) s_l1b[tid]  = l1_b[tid];
  if (tid < 16)  s_l1fb[tid] = l1f_b[tid];
  if (tid < 8)   s_outb[tid] = out_b[tid];

  // x staging: per async instruction, 16 consecutive lanes cover one row's 256B
  // contiguously (perfect coalescing). Thread t -> (row tid>>4, 16B unit tid&15);
  // 8 instructions step 8 rows each: 64 rows x 64 f32 per chunk.
  const int urow = tid >> 4;   // 0..7
  const int ucol = tid & 15;   // 0..15 (16B unit within 256B row chunk)
  const float* gxu = x + (size_t)(rowBase + urow) * L1DIM + ucol * 4;
  float* lxu0 = &xbuf[0][urow * XSTRIDE + ucol * 4];
  float* lxu1 = &xbuf[1][urow * XSTRIDE + ucol * 4];

  // A-fragment base (per lane): row (wave*16+m), K offset 16*half within chunk.
  const float* aPtr0 = &xbuf[0][(wave * 16 + m) * XSTRIDE + 16 * half];
  const float* aPtr1 = &xbuf[1][(wave * 16 + m) * XSTRIDE + 16 * half];
  // B-fragment base: weights are [N][K] f16; lane reads 32B contiguous at K = kbase + 16*half.
  const _Float16* wPtr = wf16 + (size_t)m * L1DIM + half * 16;

  const v8f vzero = {0.f, 0.f, 0.f, 0.f, 0.f, 0.f, 0.f, 0.f};
  v8f acc[NTILES];
#pragma unroll
  for (int t = 0; t < NTILES; ++t) acc[t] = vzero;

  // prologue: chunk 0 -> buf 0
#pragma unroll
  for (int i = 0; i < 8; ++i)
    cp16(gxu + (size_t)i * 8 * L1DIM, lxu0 + i * 8 * XSTRIDE);

  for (int s = 0; s < NSTEPS; ++s) {
    const int cur = s & 1;
    __syncthreads();  // next buffer no longer being read by any wave
    if (s + 1 < NSTEPS) {
      const float* gk = gxu + (size_t)(s + 1) * KC;
      float* lk = cur ? lxu0 : lxu1;
#pragma unroll
      for (int i = 0; i < 8; ++i)
        cp16(gk + (size_t)i * 8 * L1DIM, lk + i * 8 * XSTRIDE);
      wait_async<8>();  // the 8 older ops (current buffer) are done
    } else {
      wait_async<0>();
    }
    __syncthreads();  // current buffer fully in LDS for all waves

    const float* xr0 = cur ? aPtr1 : aPtr0;
    const _Float16* wk = wPtr + s * KC;
#pragma unroll
    for (int u = 0; u < 2; ++u) {  // two 16x16x32 k-tiles per staged chunk
      // Build f16 A fragment from f32 LDS (layout: element s = K 16h+s; halves 16-apart).
      const float* xr = xr0 + u * 32;
      v4f a0 = *(const v4f*)(xr);        // K = 16h+0..3
      v4f a1 = *(const v4f*)(xr + 4);    // K = 16h+4..7
      v4f a2 = *(const v4f*)(xr + 8);    // K = 16h+8..11
      v4f a3 = *(const v4f*)(xr + 12);   // K = 16h+12..15
      v16h A;
#pragma unroll
      for (int e = 0; e < 4; ++e) {
        A[e]      = (_Float16)a0[e];
        A[4 + e]  = (_Float16)a1[e];
        A[8 + e]  = (_Float16)a2[e];
        A[12 + e] = (_Float16)a3[e];
      }

      const _Float16* wku = wk + u * 32;
#pragma unroll
      for (int t = 0; t < NTILES; ++t) {
        v16h Bf = *(const v16h*)(wku + (size_t)t * 16 * L1DIM);  // 32B contiguous, L2-hot
        acc[t] = __builtin_amdgcn_wmma_f32_16x16x32_f16(false, A, false, Bf,
                                                        (short)0, acc[t], false, false);
      }
    }
  }

  // ---------------- epilogue ----------------
  // Spill this wave's 16x144 f32 tile (C layout: lane n = col, VGPR v -> row v+8*half).
  float* accW = &accbuf[wave * 16 * ACCSTRIDE];
#pragma unroll
  for (int t = 0; t < NTILES; ++t)
#pragma unroll
    for (int v = 0; v < 8; ++v)
      accW[(v + 8 * half) * ACCSTRIDE + t * 16 + m] = acc[t][v];
  // Wave-private region: same-wave LDS ops are in order; no barrier needed.

  const int lr  = m;                         // row within this wave's tile
  const int R   = rowBase + wave * 16 + lr;  // global row (both halves same row)
  const float* arow = accW + lr * ACCSTRIDE;
  const int idx  = ls_indices[R];
  const int i16  = idx * 16;
  const int i32o = idx * 32;

  const float l1c15 = arow[i16 + 15] + s_l1b[i16 + 15];
  const float l1f15 = arow[128 + 15] + s_l1fb[15];

  float l1x[30];
#pragma unroll
  for (int j = 0; j < 15; ++j) {
    const float z  = (arow[i16 + j] + s_l1b[i16 + j]) + (arow[128 + j] + s_l1fb[j]);
    const float sq = z * z * (127.0f / 128.0f);
    l1x[j]      = fminf(fmaxf(sq, 0.f), 1.f);
    l1x[15 + j] = fminf(fmaxf(z, 0.f), 1.f);
  }

  // Stage 2+3: half-wave splits the 32 outputs; reduce across halves.
  float partial = 0.f;
#pragma unroll
  for (int oo = 0; oo < 16; ++oo) {
    const int o = half * 16 + oo;
    float s2 = s_l2b[i32o + o];
    const float* wrow = &s_l2w[(i32o + o) * 30];
#pragma unroll
    for (int j = 0; j < 30; j += 2)
      s2 += l1x[j] * wrow[j] + l1x[j + 1] * wrow[j + 1];
    const float l2x = fminf(fmaxf(s2, 0.f), 1.f);
    partial += l2x * s_outw[i32o + o];
  }
  partial += __shfl_xor(partial, 16, 32);

  if (half == 0)
    out[R] = partial + s_outb[idx] + l1f15 + l1c15;
}

extern "C" void kernel_launch(void* const* d_in, const int* in_sizes, int n_in,
                              void* d_out, int out_size, void* d_ws, size_t ws_size,
                              hipStream_t stream) {
  const float* x      = (const float*)d_in[0];
  const int*   ls     = (const int*)d_in[1];
  const float* l1_w   = (const float*)d_in[2];
  const float* l1_b   = (const float*)d_in[3];
  const float* l1f_w  = (const float*)d_in[4];
  const float* l1f_b  = (const float*)d_in[5];
  const float* l2_w   = (const float*)d_in[6];
  const float* l2_b   = (const float*)d_in[7];
  const float* out_w  = (const float*)d_in[8];
  const float* out_b  = (const float*)d_in[9];
  float* out = (float*)d_out;
  _Float16* wf16 = (_Float16*)d_ws;  // 144*3072*2 = 884,736 bytes of scratch

  convert_w<<<432, 256, 0, stream>>>(l1_w, l1f_w, wf16);
  layerstacks_fused<<<BATCH / RPB, NTHREADS, 0, stream>>>(
      x, ls, wf16, l1_b, l1f_b, l2_w, l2_b, out_w, out_b, out);
}
